// TemporalCausalMaps_9620726743152
// MI455X (gfx1250) — matrix operand order repaired and checked by reading, not compiled, not run on hardware
//
#include <hip/hip_runtime.h>
#include <hip/hip_bf16.h>
#include <stddef.h>

// ---------------------------------------------------------------------------
// TemporalCausalMaps on MI455X (gfx1250, wave32).
//
// Mapping: 1 wave (32 lanes) per block, each block owns a tile of 16 samples
// and runs the ENTIRE sequential recursion (idx loop -> node chain -> GRU
// scan) privately: B=2048 -> 128 independent blocks, no global sync.
// All GEMMs are M=16(samples) x N(feat) x K(feat) via v_wmma_f32_16x16x32_bf16
// (bf16 A/B, f32 accumulate). Weights are staged to LDS as zero-padded bf16 in
// [N][Kpad] (torch [out,in]) layout so B-fragments are two 16B ds loads per lane.
// ---------------------------------------------------------------------------

#define NN    8     // nodes
#define TS    8     // time steps
#define ZD    32    // latent
#define EMB   8
#define GH    24    // gru hidden
#define HID   64
#define MS    16    // samples per block (wave)

typedef __attribute__((ext_vector_type(16))) __bf16         v16bf;
typedef __attribute__((ext_vector_type(16))) unsigned short v16us;
typedef __attribute__((ext_vector_type(8)))  unsigned short v8us;
typedef __attribute__((ext_vector_type(8)))  float          v8f;

enum { ACT_NONE = 0, ACT_RELU = 1, ACT_LEAKY = 2 };

struct Smem {
  // --- bf16 (as u16 bits) weight stages, [Npad][Kpad], K contiguous ---
  unsigned short wProj[32 * 64];     // proj_w   [24x48] -> [32][64]
  unsigned short wGP  [32 * 32];     // gru_proj [24x24] -> [32][32]
  unsigned short wIH[2][80 * 32];    // gru_wih  [72x24] -> [80][32]
  unsigned short wHH[2][80 * 32];    // gru_whh  [72x24] -> [80][32]
  unsigned short wC1 [64 * 32];      // cm_w1    [64x24] -> [64][32]
  unsigned short wC2 [64 * 64];      // cm_w2    [64x64]
  unsigned short wC3 [16 * 64];      // cm_w3    [ 1x64] -> [16][64]
  // --- bf16 activation buffers, row-major [MS][Kpad] ---
  unsigned short Gseq[TS][MS * 32];  // gru inputs / layer-2 outputs
  unsigned short Yseq[TS][MS * 32];  // layer-1 outputs
  unsigned short Xb [MS * 64];       // pipeline input (z|emb_self|emb_pred|0)
  unsigned short X2b[MS * 64];       // cm2 output
  unsigned short Tb [MS * 32];       // proj output
  unsigned short Hb [MS * 32];       // gru hidden (bf16 view)
  // --- f32 ---
  float bProj[32], bGP[32], bIH[2][80], bHH[2][80], bC1[64], bC2[64], bC3[16];
  float cw1s[2 * 32], cb1s[2 * 32], cw2s[2 * 256], cb2s[2 * 8];  // cond nets (self,pred)
  float dec[NN][MS * TS];            // decoded scalars: dec[node][m*8+t]
  float GI[MS * 80], GH_[MS * 80];   // gru gate pre-activations
  float Hf[MS * 32];                 // gru hidden (f32 master copy)
};

__device__ __forceinline__ unsigned short f2bf(float f) {
  unsigned u = __float_as_uint(f);
  u += 0x7FFFu + ((u >> 16) & 1u);   // round-to-nearest-even
  return (unsigned short)(u >> 16);
}

// A-fragment (16x32 bf16): lane l -> row l%16; halves 0..7 = K (l/16)*8..,
// halves 8..15 = K 16+(l/16)*8..  (per CDNA5 ISA 16-bit A table)
__device__ __forceinline__ v16bf ld_A(const unsigned short* X, int ldk, int kc, int lane) {
  int m = lane & 15, g = (lane >> 4) & 1;
  const unsigned short* p0 = X + m * ldk + kc + g * 8;
  v8us a0 = *(const v8us*)p0;
  v8us a1 = *(const v8us*)(p0 + 16);
  v16us t;
#pragma unroll
  for (int i = 0; i < 8; ++i) { t[i] = a0[i]; t[8 + i] = a1[i]; }
  return __builtin_bit_cast(v16bf, t);
}

// B-fragment (32x16 bf16), W in LDS as [Npad][Kpad] (K contiguous):
// lane l -> column ntile*16 + l%16, halves 0..7 = K kc+(l/16)*8.., halves 8..15 = K kc+16+(l/16)*8..
__device__ __forceinline__ v16bf ld_B(const unsigned short* W, int ldk, int nt, int kc, int lane) {
  int n = nt * 16 + (lane & 15), g = (lane >> 4) & 1;
  const unsigned short* p0 = W + n * ldk + kc + g * 8;
  v8us b0 = *(const v8us*)p0;
  v8us b1 = *(const v8us*)(p0 + 16);
  v16us t;
#pragma unroll
  for (int i = 0; i < 8; ++i) { t[i] = b0[i]; t[8 + i] = b1[i]; }
  return __builtin_bit_cast(v16bf, t);
}

__device__ __forceinline__ v8f wmma_bf16(v16bf a, v16bf b, v8f c) {
  return __builtin_amdgcn_wmma_f32_16x16x32_bf16(false, a, false, b, (short)0, c, false, false);
}

__device__ __forceinline__ v8f bias_frag(const float* b, int nt, int lane) {
  float f = b[nt * 16 + (lane & 15)];
  v8f c = {f, f, f, f, f, f, f, f};
  return c;
}

// D (v8f): lane l, vgpr v -> row v + 8*(l/16), col nt*16 + l%16
__device__ __forceinline__ void store_d_bf16(unsigned short* dst, int ldn, int nt, v8f c,
                                             int act, int lane) {
  int n = nt * 16 + (lane & 15), g = (lane >> 4) & 1;
#pragma unroll
  for (int v = 0; v < 8; ++v) {
    float f = c[v];
    if (act == ACT_RELU)       f = fmaxf(f, 0.f);
    else if (act == ACT_LEAKY) f = (f >= 0.f) ? f : 0.01f * f;
    dst[(v + 8 * g) * ldn + n] = f2bf(f);
  }
}

__device__ __forceinline__ void store_d_f32(float* dst, int ldn, int nt, v8f c, int lane) {
  int n = nt * 16 + (lane & 15), g = (lane >> 4) & 1;
#pragma unroll
  for (int v = 0; v < 8; ++v) dst[(v + 8 * g) * ldn + n] = c[v];
}

__device__ __forceinline__ void stage_w(const float* src, int N, int K,
                                        unsigned short* dst, int Np, int Kp, int lane) {
  for (int i = lane; i < Np * Kp; i += 32) {
    int n = i / Kp, k = i - n * Kp;
    dst[i] = f2bf((n < N && k < K) ? src[n * K + k] : 0.f);
  }
}
__device__ __forceinline__ void stage_b(const float* src, int N, float* dst, int Np, int lane) {
  for (int i = lane; i < Np; i += 32) dst[i] = (i < N) ? src[i] : 0.f;
}

__device__ __forceinline__ float sigm(float x) { return 1.f / (1.f + __expf(-x)); }

// cond_net: relu(x*w1+b1) @ w2[e].T + b2[e]   (1 -> 32 -> 1 output channel e)
__device__ __forceinline__ float cond_eval(const float* w1, const float* b1,
                                           const float* w2, const float* b2,
                                           float x, int e) {
  float acc = b2[e];
#pragma unroll 8
  for (int j = 0; j < 32; ++j) {
    float h = fmaxf(fmaf(x, w1[j], b1[j]), 0.f);
    acc = fmaf(h, w2[e * 32 + j], acc);
  }
  return acc;
}

// One torch-GRU layer over L steps: in/out seqs are [t][MS*32] bf16 (cols>=24 zero)
__device__ __forceinline__ void gru_layer(Smem& S, int layer,
                                          const unsigned short* inSeq,
                                          unsigned short* outSeq, int L, int lane) {
  for (int i = lane; i < MS * 32; i += 32) { S.Hf[i] = 0.f; S.Hb[i] = 0; }
  __syncthreads();
  const unsigned short* wih = S.wIH[layer];
  const unsigned short* whh = S.wHH[layer];
  const float* bih = S.bIH[layer];
  const float* bhh = S.bHH[layer];
  for (int t = 0; t < L; ++t) {
    v16bf ax = ld_A(inSeq + t * MS * 32, 32, 0, lane);
    v16bf ah = ld_A(S.Hb, 32, 0, lane);
#pragma unroll
    for (int nt = 0; nt < 5; ++nt) {   // 72 gates -> 5 N-tiles of 16
      v8f ci = bias_frag(bih, nt, lane);
      ci = wmma_bf16(ax, ld_B(wih, 32, nt, 0, lane), ci);
      store_d_f32(S.GI, 80, nt, ci, lane);
      v8f ch = bias_frag(bhh, nt, lane);
      ch = wmma_bf16(ah, ld_B(whh, 32, nt, 0, lane), ch);
      store_d_f32(S.GH_, 80, nt, ch, lane);
    }
    __syncthreads();
    for (int i = lane; i < MS * GH; i += 32) {  // gate order r, z, n
      int m = i / GH, j = i - GH * m;
      float r  = sigm(S.GI[m * 80 +      j] + S.GH_[m * 80 +      j]);
      float zg = sigm(S.GI[m * 80 + 24 + j] + S.GH_[m * 80 + 24 + j]);
      float nn = tanhf(S.GI[m * 80 + 48 + j] + r * S.GH_[m * 80 + 48 + j]);
      float h  = (1.f - zg) * nn + zg * S.Hf[m * 32 + j];
      S.Hf[m * 32 + j] = h;
      unsigned short hb = f2bf(h);
      S.Hb[m * 32 + j] = hb;
      outSeq[t * MS * 32 + m * 32 + j] = hb;
    }
    for (int i = lane; i < MS * 8; i += 32) {   // keep pad cols zero
      int m = i >> 3, k = i & 7;
      outSeq[t * MS * 32 + m * 32 + 24 + k] = 0;
    }
    __syncthreads();
  }
}

__global__ void __launch_bounds__(32)
tcm_kernel(const float* __restrict__ z,
           const float* __restrict__ cond_w1, const float* __restrict__ cond_b1,
           const float* __restrict__ cond_w2, const float* __restrict__ cond_b2,
           const float* __restrict__ proj_w,  const float* __restrict__ proj_b,
           const float* __restrict__ gru_proj_w, const float* __restrict__ gru_proj_b,
           const float* __restrict__ gru_wih, const float* __restrict__ gru_whh,
           const float* __restrict__ gru_bih, const float* __restrict__ gru_bhh,
           const float* __restrict__ cm_w1, const float* __restrict__ cm_b1,
           const float* __restrict__ cm_w2, const float* __restrict__ cm_b2,
           const float* __restrict__ cm_w3, const float* __restrict__ cm_b3,
           float* __restrict__ out, int Bsz) {
  extern __shared__ __align__(128) char smem_raw[];
  Smem& S = *reinterpret_cast<Smem*>(smem_raw);
  const int lane = threadIdx.x;
  const int b0 = blockIdx.x * MS;

  // gru_proj is node-independent: stage once
  stage_w(gru_proj_w, GH, GH, S.wGP, 32, 32, lane);
  stage_b(gru_proj_b, GH, S.bGP, 32, lane);
  __syncthreads();

  for (int idx = 0; idx < TS; ++idx) {
    const int L = idx + 1;
    for (int node = 0; node < NN; ++node) {
      // ---------------- stage this node's weights to LDS (bf16) -------------
      stage_w(proj_w + node * 24 * 48, 24, 48, S.wProj, 32, 64, lane);
      stage_b(proj_b + node * 24, 24, S.bProj, 32, lane);
#pragma unroll
      for (int l = 0; l < 2; ++l) {
        stage_w(gru_wih + (node * 2 + l) * 72 * 24, 72, 24, S.wIH[l], 80, 32, lane);
        stage_w(gru_whh + (node * 2 + l) * 72 * 24, 72, 24, S.wHH[l], 80, 32, lane);
        stage_b(gru_bih + (node * 2 + l) * 72, 72, S.bIH[l], 80, lane);
        stage_b(gru_bhh + (node * 2 + l) * 72, 72, S.bHH[l], 80, lane);
      }
      stage_w(cm_w1 + node * 64 * 24, 64, 24, S.wC1, 64, 32, lane);
      stage_w(cm_w2 + node * 64 * 64, 64, 64, S.wC2, 64, 64, lane);
      stage_w(cm_w3 + node * 64,       1, 64, S.wC3, 16, 64, lane);
      stage_b(cm_b1 + node * 64, 64, S.bC1, 64, lane);
      stage_b(cm_b2 + node * 64, 64, S.bC2, 64, lane);
      stage_b(cm_b3 + node,       1, S.bC3, 16, lane);
      int pn = (node > 0) ? node - 1 : node;
      for (int i = lane; i < 32; i += 32) {
        S.cw1s[i] = cond_w1[node * 32 + i]; S.cw1s[32 + i] = cond_w1[pn * 32 + i];
        S.cb1s[i] = cond_b1[node * 32 + i]; S.cb1s[32 + i] = cond_b1[pn * 32 + i];
      }
      for (int i = lane; i < 256; i += 32) {
        S.cw2s[i] = cond_w2[node * 256 + i]; S.cw2s[256 + i] = cond_w2[pn * 256 + i];
      }
      for (int i = lane; i < 8; i += 32) {
        S.cb2s[i] = cond_b2[node * 8 + i]; S.cb2s[8 + i] = cond_b2[pn * 8 + i];
      }
      __syncthreads();

      // ---- Phase A: build inputs, proj -> leaky -> gru_proj -> Gseq[t] -----
      for (int t = 0; t < L; ++t) {
        for (int i = lane; i < MS * ZD; i += 32) {   // z part, cols 0..31
          int m = i >> 5, k = i & 31;
          S.Xb[m * 64 + k] =
              f2bf(z[(((size_t)node * Bsz + b0 + m) * TS + t) * ZD + k]);
        }
        if (t + 1 < L)
          __builtin_prefetch(&z[(((size_t)node * Bsz + b0 + (lane & 15)) * TS + t + 1) * ZD], 0, 1);
        const bool haveSelf = (idx > 0) && (t < idx);
        const bool havePred = (node > 0) && (idx > 0);
        for (int i = lane; i < MS * EMB; i += 32) {  // embeddings, cols 32..47
          int m = i >> 3, e = i & 7;
          float vs = 0.f, vp = 0.f;
          if (haveSelf)
            vs = cond_eval(S.cw1s, S.cb1s, S.cw2s, S.cb2s, S.dec[node][m * TS + t], e);
          if (havePred)
            vp = cond_eval(S.cw1s + 32, S.cb1s + 32, S.cw2s + 256, S.cb2s + 8,
                           S.dec[node - 1][m * TS + t], e);
          S.Xb[m * 64 + 32 + e] = f2bf(vs);
          S.Xb[m * 64 + 40 + e] = f2bf(vp);
        }
        for (int i = lane; i < MS * 16; i += 32) {   // pad cols 48..63
          int m = i >> 4, k = i & 15;
          S.Xb[m * 64 + 48 + k] = 0;
        }
        __syncthreads();
        {  // proj: [16x48]@[48x24] -> leaky -> Tb
          v16bf a0 = ld_A(S.Xb, 64, 0, lane), a1 = ld_A(S.Xb, 64, 32, lane);
#pragma unroll
          for (int nt = 0; nt < 2; ++nt) {
            v8f c = bias_frag(S.bProj, nt, lane);
            c = wmma_bf16(a0, ld_B(S.wProj, 64, nt, 0, lane), c);
            c = wmma_bf16(a1, ld_B(S.wProj, 64, nt, 32, lane), c);
            store_d_bf16(S.Tb, 32, nt, c, ACT_LEAKY, lane);
          }
        }
        __syncthreads();
        {  // gru_proj: [16x24]@[24x24] -> Gseq[t]
          v16bf a = ld_A(S.Tb, 32, 0, lane);
#pragma unroll
          for (int nt = 0; nt < 2; ++nt) {
            v8f c = bias_frag(S.bGP, nt, lane);
            c = wmma_bf16(a, ld_B(S.wGP, 32, nt, 0, lane), c);
            store_d_bf16(S.Gseq[t], 32, nt, c, ACT_NONE, lane);
          }
        }
        __syncthreads();
      }

      // ---- Phase B/C: two GRU layers (sequential scan over t) --------------
      gru_layer(S, 0, &S.Gseq[0][0], &S.Yseq[0][0], L, lane);
      gru_layer(S, 1, &S.Yseq[0][0], &S.Gseq[0][0], L, lane);

      // ---- Phase D: causal map 24->64->64->1 per position ------------------
      for (int t = 0; t < L; ++t) {
        {  // cm1
          v16bf a = ld_A(S.Gseq[t], 32, 0, lane);
#pragma unroll
          for (int nt = 0; nt < 4; ++nt) {
            v8f c = bias_frag(S.bC1, nt, lane);
            c = wmma_bf16(a, ld_B(S.wC1, 32, nt, 0, lane), c);
            store_d_bf16(S.Xb, 64, nt, c, ACT_RELU, lane);
          }
        }
        __syncthreads();
        {  // cm2
          v16bf a0 = ld_A(S.Xb, 64, 0, lane), a1 = ld_A(S.Xb, 64, 32, lane);
#pragma unroll
          for (int nt = 0; nt < 4; ++nt) {
            v8f c = bias_frag(S.bC2, nt, lane);
            c = wmma_bf16(a0, ld_B(S.wC2, 64, nt, 0, lane), c);
            c = wmma_bf16(a1, ld_B(S.wC2, 64, nt, 32, lane), c);
            store_d_bf16(S.X2b, 64, nt, c, ACT_RELU, lane);
          }
        }
        __syncthreads();
        {  // cm3 (N=1, column 0 of tile 0)
          v16bf a0 = ld_A(S.X2b, 64, 0, lane), a1 = ld_A(S.X2b, 64, 32, lane);
          v8f c = bias_frag(S.bC3, 0, lane);
          c = wmma_bf16(a0, ld_B(S.wC3, 64, 0, 0, lane), c);
          c = wmma_bf16(a1, ld_B(S.wC3, 64, 0, 32, lane), c);
          if ((lane & 15) == 0) {
            int g = (lane >> 4) & 1;
#pragma unroll
            for (int v = 0; v < 8; ++v) S.dec[node][(v + 8 * g) * TS + t] = c[v];
          }
        }
        __syncthreads();
      }
      // emit out[b, idx, node] = decoded value at position idx
      if (lane < MS)
        out[((size_t)(b0 + lane)) * (TS * NN) + idx * NN + node] =
            S.dec[node][lane * TS + idx];
      __syncthreads();
    }
  }
}

extern "C" void kernel_launch(void* const* d_in, const int* in_sizes, int n_in,
                              void* d_out, int out_size, void* d_ws, size_t ws_size,
                              hipStream_t stream) {
  (void)n_in; (void)d_ws; (void)ws_size; (void)out_size;
  const float* z          = (const float*)d_in[0];
  const float* cond_w1    = (const float*)d_in[1];
  const float* cond_b1    = (const float*)d_in[2];
  const float* cond_w2    = (const float*)d_in[3];
  const float* cond_b2    = (const float*)d_in[4];
  const float* proj_w     = (const float*)d_in[5];
  const float* proj_b     = (const float*)d_in[6];
  const float* gru_proj_w = (const float*)d_in[7];
  const float* gru_proj_b = (const float*)d_in[8];
  const float* gru_wih    = (const float*)d_in[9];
  const float* gru_whh    = (const float*)d_in[10];
  const float* gru_bih    = (const float*)d_in[11];
  const float* gru_bhh    = (const float*)d_in[12];
  const float* cm_w1      = (const float*)d_in[13];
  const float* cm_b1      = (const float*)d_in[14];
  const float* cm_w2      = (const float*)d_in[15];
  const float* cm_b2      = (const float*)d_in[16];
  const float* cm_w3      = (const float*)d_in[17];
  const float* cm_b3      = (const float*)d_in[18];
  // d_in[19] = end_wind (known constant 8 per reference)

  int Bsz = in_sizes[0] / (NN * TS * ZD);   // 2048
  int nBlocks = Bsz / MS;                   // 128 independent batch tiles
  tcm_kernel<<<dim3(nBlocks), dim3(32), sizeof(Smem), stream>>>(
      z, cond_w1, cond_b1, cond_w2, cond_b2, proj_w, proj_b,
      gru_proj_w, gru_proj_b, gru_wih, gru_whh, gru_bih, gru_bhh,
      cm_w1, cm_b1, cm_w2, cm_b2, cm_w3, cm_b3, (float*)d_out, Bsz);
}
